// RWKV_SpatialMix_1D_35759897706978
// MI455X (gfx1250) — compile-verified, matrix operand order checked by reading, not run on hardware
//
#include <hip/hip_runtime.h>

// ---------------------------------------------------------------------------
// RWKV spatial mix for MI455X (gfx1250, wave32, WMMA)
//   B=8, T=4096, C=1024. 4 GEMMs (bf16 WMMA, f32 accum) + chunked WKV scan.
//   GEMM: 128x256 block tile, 8 waves, each wave 4x4 WMMA 16x16 tiles.
// ---------------------------------------------------------------------------

typedef __bf16 bf16;
typedef __attribute__((ext_vector_type(8)))  float v8f;
typedef __attribute__((ext_vector_type(8)))  bf16  v8bf;
typedef __attribute__((ext_vector_type(16))) bf16  v16bf;

constexpr int Bd = 8;
constexpr int Td = 4096;
constexpr int Cd = 1024;
constexpr int Md = Bd * Td;          // 32768 rows
constexpr int SCH = 16;              // scan chunks
constexpr int CHL = Td / SCH;        // 256 steps per chunk

// ---------------------------------------------------------------------------
// fp32 -> bf16 matrix convert (weights)
// ---------------------------------------------------------------------------
__global__ __launch_bounds__(256) void cvt_bf16_kernel(const float* __restrict__ src,
                                                       bf16* __restrict__ dst, int n) {
    int i = (blockIdx.x * 256 + threadIdx.x) * 4;
    if (i < n) {
        float4 q = *(const float4*)(src + i);
        dst[i + 0] = (bf16)q.x;
        dst[i + 1] = (bf16)q.y;
        dst[i + 2] = (bf16)q.z;
        dst[i + 3] = (bf16)q.w;
    }
}

// ---------------------------------------------------------------------------
// 128x256 block-tile GEMM:  out[M,N] = A[M,K] * W^T  (W is N x K row-major)
//   AMODE 1: A is built on the fly from fp32 x via q-shift + token mix (bf16)
//   AMODE 0: A is a preformed bf16 matrix
//   EPI   1: apply sigmoid in epilogue (for r -> sr)
// 8 waves/block (2 M x 4 N), each wave computes 4x4 WMMA 16x16 tiles
// (128 f32 acc VGPRs/lane). Per K-step per wave: 16 ds_load_b128 : 16 WMMA.
// LDS: double-buffered 128x32 (A) and 256x32 (B) bf16 tiles = 48 KB.
// ---------------------------------------------------------------------------
template <int AMODE, int EPI>
__global__ __launch_bounds__(256) void gemm128_kernel(const float* __restrict__ X,
                                                      const bf16* __restrict__ A16,
                                                      const float* __restrict__ mix,
                                                      const bf16* __restrict__ W16,
                                                      float* __restrict__ out) {
    __shared__ bf16 As[2][128 * 32];
    __shared__ bf16 Bs[2][256 * 32];

    const int tid   = threadIdx.x;
    const int mBase = blockIdx.x * 128;
    const int nBase = blockIdx.y * 256;

    // A loader: 2 threads per tile row, 16 bf16 elements each
    const int arow = tid >> 1;
    const int aseg = (tid & 1) * 16;
    // B loader: 1 thread per tile row, 32 bf16 elements (64 B)

    // wave / lane mapping for WMMA fragments (wave32)
    const int lane = tid & 31;
    const int r    = lane & 15;
    const int half = lane >> 4;
    const int w    = tid >> 5;
    const int wm   = w & 1;   // 2 M sub-tiles of 64 rows
    const int wn   = w >> 1;  // 4 N sub-tiles of 64 cols

    v8f acc[4][4] = {};

    v8bf raA0, raA1;      // A staging (16 bf16)
    v8bf raB[4];          // B staging (32 bf16)

    auto loadStage = [&](int kt) {
        const int c0 = kt * 32 + aseg;  // K-column base of A 16-segment
        // -------- A tile (128 x 32) --------
        if constexpr (AMODE == 1) {
            const int grow = mBase + arow;
            const int t    = grow & (Td - 1);
            int srow;
            if (c0 < 256)       srow = (t > 0) ? grow - 1 : -1;       // fwd shift
            else if (c0 < 512)  srow = (t < Td - 1) ? grow + 1 : -1;  // bwd shift
            else                srow = grow;                          // pass-through
            float rx[16], rxx[16], rm[16];
            const float4* xp4 = (const float4*)(X + (size_t)grow * Cd + c0);
            #pragma unroll
            for (int j = 0; j < 4; ++j) {
                float4 q = xp4[j];
                rx[4*j+0]=q.x; rx[4*j+1]=q.y; rx[4*j+2]=q.z; rx[4*j+3]=q.w;
            }
            if (srow >= 0) {
                const float4* sp4 = (const float4*)(X + (size_t)srow * Cd + c0);
                #pragma unroll
                for (int j = 0; j < 4; ++j) {
                    float4 q = sp4[j];
                    rxx[4*j+0]=q.x; rxx[4*j+1]=q.y; rxx[4*j+2]=q.z; rxx[4*j+3]=q.w;
                }
            } else {
                #pragma unroll
                for (int j = 0; j < 16; ++j) rxx[j] = 0.0f;
            }
            const float4* mp4 = (const float4*)(mix + c0);
            #pragma unroll
            for (int j = 0; j < 4; ++j) {
                float4 q = mp4[j];
                rm[4*j+0]=q.x; rm[4*j+1]=q.y; rm[4*j+2]=q.z; rm[4*j+3]=q.w;
            }
            #pragma unroll
            for (int j = 0; j < 8; ++j) {
                raA0[j] = (bf16)(rx[j]     * rm[j]     + rxx[j]     * (1.0f - rm[j]));
                raA1[j] = (bf16)(rx[j + 8] * rm[j + 8] + rxx[j + 8] * (1.0f - rm[j + 8]));
            }
        } else {
            const int grow = mBase + arow;
            const bf16* ap = A16 + (size_t)grow * Cd + c0;
            raA0 = *(const v8bf*)ap;
            raA1 = *(const v8bf*)(ap + 8);
        }
        // -------- W tile (256 x 32): one 64 B row per thread --------
        const int nrow = nBase + tid;
        const bf16* wp = W16 + (size_t)nrow * Cd + kt * 32;
        #pragma unroll
        for (int j = 0; j < 4; ++j) raB[j] = *(const v8bf*)(wp + 8 * j);
        // hint next K-step into cache (emits global_prefetch_b8)
        __builtin_prefetch((const void*)(wp + 32), 0, 3);
    };

    auto storeStage = [&](int buf) {
        *(v8bf*)&As[buf][arow * 32 + aseg]     = raA0;
        *(v8bf*)&As[buf][arow * 32 + aseg + 8] = raA1;
        #pragma unroll
        for (int j = 0; j < 4; ++j) *(v8bf*)&Bs[buf][tid * 32 + 8 * j] = raB[j];
    };

    loadStage(0);
    storeStage(0);
    __syncthreads();

    for (int kt = 0; kt < 32; ++kt) {
        const int cur = kt & 1;
        if (kt + 1 < 32) loadStage(kt + 1);

        // A fragments: lane (r, half) holds row M=r, K = half*8..+7 and 16+half*8..+7
        v16bf afrag[4];
        #pragma unroll
        for (int mt = 0; mt < 4; ++mt) {
            const bf16* pa = &As[cur][(wm * 64 + mt * 16 + r) * 32 + half * 8];
            v8bf lo = *(const v8bf*)pa;
            v8bf hi = *(const v8bf*)(pa + 16);
            afrag[mt] = __builtin_shufflevector(lo, hi, 0,1,2,3,4,5,6,7,8,9,10,11,12,13,14,15);
        }
        #pragma unroll
        for (int nt = 0; nt < 4; ++nt) {
            const bf16* pb = &Bs[cur][(wn * 64 + nt * 16 + r) * 32 + half * 8];
            v8bf lo = *(const v8bf*)pb;
            v8bf hi = *(const v8bf*)(pb + 16);
            v16bf bfrag = __builtin_shufflevector(lo, hi, 0,1,2,3,4,5,6,7,8,9,10,11,12,13,14,15);
            #pragma unroll
            for (int mt = 0; mt < 4; ++mt) {
                acc[mt][nt] = __builtin_amdgcn_wmma_f32_16x16x32_bf16(
                    false, afrag[mt], false, bfrag, (short)0, acc[mt][nt], false, false);
            }
        }

        if (kt + 1 < 32) storeStage(cur ^ 1);
        __syncthreads();
    }

    // epilogue: C/D layout -> VGPR q holds row (q + half*8), col r
    #pragma unroll
    for (int mt = 0; mt < 4; ++mt) {
        const int m0 = mBase + wm * 64 + mt * 16 + half * 8;
        #pragma unroll
        for (int nt = 0; nt < 4; ++nt) {
            const int n = nBase + wn * 64 + nt * 16 + r;
            #pragma unroll
            for (int q = 0; q < 8; ++q) {
                float vv = acc[mt][nt][q];
                if constexpr (EPI == 1) vv = 1.0f / (1.0f + __expf(-vv));
                out[(size_t)(m0 + q) * Cd + n] = vv;
            }
        }
    }
}

// ---------------------------------------------------------------------------
// WKV scan, chunked 3-pass (affine recurrence with constant multiplier exp(w)):
//   state s = (a,b);  s_t = s_{t-1}*ew + (ek*v, ek)
// Pass 1: per-(b,c,chunk) aggregate assuming zero start  (131072 threads)
// Pass 2: per-(b,c) combine 16 chunk aggregates -> chunk start states
// Pass 3: per-(b,c,chunk) replay chunk, emit y = sigmoid(r)*wkv as bf16
// ---------------------------------------------------------------------------
__global__ __launch_bounds__(256) void wkv_pass1_kernel(const float* __restrict__ kb,
                                                        const float* __restrict__ vb,
                                                        const float* __restrict__ decay,
                                                        float2* __restrict__ aggs) {
    int g = blockIdx.x * 256 + threadIdx.x;       // B*C*SCH threads
    int c = g % Cd;
    int chunk = (g / Cd) % SCH;
    int b = g / (Cd * SCH);
    float ew = __expf(decay[c] * (1.0f / (float)Td));
    size_t base = ((size_t)b * Td + (size_t)chunk * CHL) * Cd + c;
    float a = 0.0f, bb = 0.0f;
    for (int i = 0; i < CHL; ++i) {
        size_t idx = base + (size_t)i * Cd;
        float kt = kb[idx];
        float vt = vb[idx];
        float ek = __expf(kt);
        a  = a  * ew + ek * vt;
        bb = bb * ew + ek;
    }
    float2 o; o.x = a; o.y = bb;
    aggs[(size_t)(b * SCH + chunk) * Cd + c] = o;
}

__global__ __launch_bounds__(256) void wkv_pass2_kernel(const float2* __restrict__ aggs,
                                                        float2* __restrict__ starts,
                                                        const float* __restrict__ decay) {
    int g = blockIdx.x * 256 + threadIdx.x;       // B*C threads
    int c = g % Cd;
    int b = g / Cd;
    float ewL = __expf(decay[c] * ((float)CHL / (float)Td));
    float a = 0.0f, bb = 0.0f;
    for (int j = 0; j < SCH; ++j) {
        size_t idx = (size_t)(b * SCH + j) * Cd + c;
        float2 s; s.x = a; s.y = bb;
        starts[idx] = s;
        float2 ag = aggs[idx];
        a  = a  * ewL + ag.x;
        bb = bb * ewL + ag.y;
    }
}

__global__ __launch_bounds__(256) void wkv_pass3_kernel(const float* __restrict__ kb,
                                                        const float* __restrict__ vb,
                                                        const float* __restrict__ srb,
                                                        const float* __restrict__ decay,
                                                        const float* __restrict__ first,
                                                        const float2* __restrict__ starts,
                                                        bf16* __restrict__ y) {
    int g = blockIdx.x * 256 + threadIdx.x;       // B*C*SCH threads
    int c = g % Cd;
    int chunk = (g / Cd) % SCH;
    int b = g / (Cd * SCH);
    float ew = __expf(decay[c] * (1.0f / (float)Td));
    float u  = first[c] * (1.0f / (float)Td);
    float2 s = starts[(size_t)(b * SCH + chunk) * Cd + c];
    float a = s.x, bb = s.y;
    size_t base = ((size_t)b * Td + (size_t)chunk * CHL) * Cd + c;
    for (int i = 0; i < CHL; ++i) {
        size_t idx = base + (size_t)i * Cd;
        float kt = kb[idx];
        float vt = vb[idx];
        float eku = __expf(u + kt);
        float out = (a + eku * vt) / (bb + eku);
        y[idx] = (bf16)(out * srb[idx]);
        float ek = __expf(kt);
        a  = a  * ew + ek * vt;
        bb = bb * ew + ek;
    }
}

// ---------------------------------------------------------------------------
// launch
// ---------------------------------------------------------------------------
extern "C" void kernel_launch(void* const* d_in, const int* in_sizes, int n_in,
                              void* d_out, int out_size, void* d_ws, size_t ws_size,
                              hipStream_t stream) {
    const float* x     = (const float*)d_in[0];
    const float* Wk    = (const float*)d_in[1];
    const float* Wv    = (const float*)d_in[2];
    const float* Wr    = (const float*)d_in[3];
    const float* Wo    = (const float*)d_in[4];
    const float* decay = (const float*)d_in[5];
    const float* first = (const float*)d_in[6];
    const float* mix_k = (const float*)d_in[7];
    const float* mix_v = (const float*)d_in[8];
    const float* mix_r = (const float*)d_in[9];
    float* out = (float*)d_out;

    // workspace layout
    char* wsp = (char*)d_ws;
    bf16* Wk16 = (bf16*)wsp;                         // 4 x 2MB bf16 weights
    bf16* Wv16 = Wk16 + (size_t)Cd * Cd;
    bf16* Wr16 = Wv16 + (size_t)Cd * Cd;
    bf16* Wo16 = Wr16 + (size_t)Cd * Cd;
    float* kbuf  = (float*)(wsp + (size_t)8 * 1024 * 1024);   // 128MB each
    float* vbuf  = kbuf + (size_t)Md * Cd;
    float* srbuf = vbuf + (size_t)Md * Cd;
    bf16* ybuf   = (bf16*)(srbuf + (size_t)Md * Cd);          // 64MB
    float2* aggs   = (float2*)(ybuf + (size_t)Md * Cd);       // 1MB
    float2* starts = aggs + (size_t)Bd * Cd * SCH;            // 1MB

    // 1) weights -> bf16
    cvt_bf16_kernel<<<Cd * Cd / 1024, 256, 0, stream>>>(Wk, Wk16, Cd * Cd);
    cvt_bf16_kernel<<<Cd * Cd / 1024, 256, 0, stream>>>(Wv, Wv16, Cd * Cd);
    cvt_bf16_kernel<<<Cd * Cd / 1024, 256, 0, stream>>>(Wr, Wr16, Cd * Cd);
    cvt_bf16_kernel<<<Cd * Cd / 1024, 256, 0, stream>>>(Wo, Wo16, Cd * Cd);

    // 2) fused (q_shift + mix + GEMM) for k, v, and sigmoid(r)
    dim3 ggrid(Md / 128, Cd / 256);
    gemm128_kernel<1, 0><<<ggrid, 256, 0, stream>>>(x, nullptr, mix_k, Wk16, kbuf);
    gemm128_kernel<1, 0><<<ggrid, 256, 0, stream>>>(x, nullptr, mix_v, Wv16, vbuf);
    gemm128_kernel<1, 1><<<ggrid, 256, 0, stream>>>(x, nullptr, mix_r, Wr16, srbuf);

    // 3) chunked wkv scan -> y = sigmoid(r) * wkv (bf16)
    wkv_pass1_kernel<<<Bd * Cd * SCH / 256, 256, 0, stream>>>(kbuf, vbuf, decay, aggs);
    wkv_pass2_kernel<<<Bd * Cd / 256, 256, 0, stream>>>(aggs, starts, decay);
    wkv_pass3_kernel<<<Bd * Cd * SCH / 256, 256, 0, stream>>>(kbuf, vbuf, srbuf, decay,
                                                              first, starts, ybuf);

    // 4) output projection
    gemm128_kernel<0, 0><<<ggrid, 256, 0, stream>>>(nullptr, ybuf, nullptr, Wo16, out);
}
